// AvgTransformer_50259707298335
// MI455X (gfx1250) — matrix operand order repaired and checked by reading, not compiled, not run on hardware
//
#include <hip/hip_runtime.h>
#include <hip/hip_bf16.h>

typedef __attribute__((ext_vector_type(16))) _Float16 v16h;
typedef __attribute__((ext_vector_type(8)))  _Float16 v8h;
typedef __attribute__((ext_vector_type(8)))  float    v8f;

#define B_   1024
#define L_   128
#define DJ_  48
#define DW_  1024
#define DE_  1024
#define DT_  1024
#define K_   (DJ_ + DW_ + DE_)   // 2096
#define KP_  2112                // padded to multiple of 32

// ---------------- workspace layout (256B aligned regions) ----------------
#define OFF_NZ   0u
#define OFF_CJ   256u
#define OFF_CW   (OFF_CJ + (unsigned)(B_*DJ_*4))          // 196864
#define OFF_CE   (OFF_CW + (unsigned)(B_*DW_*4))
#define OFF_AH   (OFF_CE + (unsigned)(B_*DE_*4))
#define OFF_WH   (OFF_AH + (unsigned)(B_*KP_*2))

__global__ void init_nz_kernel(int* nz) {
    if (threadIdx.x < 3) nz[threadIdx.x] = 0;
}

// One block per batch item: colsum[b,d] = sum_l x[b,l,d]; also per-item total
// -> atomic count of items with nonzero total (the reference's `nz`).
// Reads stream full rows contiguously (coalesced across lanes).
__global__ __launch_bounds__(256)
void colsum_kernel(const float* __restrict__ x, float* __restrict__ colsum,
                   int* __restrict__ nz, int D) {
    const int b = blockIdx.x;
    const float* xb = x + (size_t)b * L_ * D;

    float acc[4] = {0.f, 0.f, 0.f, 0.f};
    int   didx[4];
    int   nd = 0;
    for (int d = threadIdx.x; d < D && nd < 4; d += blockDim.x) didx[nd++] = d;

    for (int l = 0; l < L_; ++l) {
        const float* row = xb + (size_t)l * D;
        for (int i = 0; i < nd; ++i) acc[i] += row[didx[i]];
    }

    float ttotal = 0.f;
    for (int i = 0; i < nd; ++i) {
        colsum[(size_t)b * D + didx[i]] = acc[i];
        ttotal += acc[i];
    }

    __shared__ float red[256];
    red[threadIdx.x] = ttotal;
    __syncthreads();
    for (int off = blockDim.x >> 1; off > 0; off >>= 1) {
        if ((int)threadIdx.x < off) red[threadIdx.x] += red[threadIdx.x + off];
        __syncthreads();
    }
    if (threadIdx.x == 0 && red[0] != 0.0f) atomicAdd(nz, 1);
}

// Pack h = concat(j,w,e)/nz into f16 [B, KP_] with zero padding for K>=2096.
__global__ __launch_bounds__(256)
void pack_A_kernel(const float* __restrict__ cj, const float* __restrict__ cw,
                   const float* __restrict__ ce, const int* __restrict__ nz,
                   _Float16* __restrict__ Ah) {
    const int b = blockIdx.x;
    const float ij = 1.0f / fmaxf((float)nz[0], 1.0f);
    const float iw = 1.0f / fmaxf((float)nz[1], 1.0f);
    const float ie = 1.0f / fmaxf((float)nz[2], 1.0f);
    _Float16* Ab = Ah + (size_t)b * KP_;
    for (int k = threadIdx.x; k < KP_; k += blockDim.x) {
        float v;
        if (k < DJ_)               v = cj[(size_t)b * DJ_ + k] * ij;
        else if (k < DJ_ + DW_)    v = cw[(size_t)b * DW_ + (k - DJ_)] * iw;
        else if (k < K_)           v = ce[(size_t)b * DE_ + (k - DJ_ - DW_)] * ie;
        else                       v = 0.0f;
        Ab[k] = (_Float16)v;
    }
}

// Pack W [DT_, K_] f32 -> f16 [DT_, KP_] (row-major over K, zero padded).
__global__ __launch_bounds__(256)
void pack_W_kernel(const float* __restrict__ W, _Float16* __restrict__ Wh) {
    const int t = blockIdx.x;
    _Float16* Wt = Wh + (size_t)t * KP_;
    const float* Ws = W + (size_t)t * K_;
    for (int k = threadIdx.x; k < KP_; k += blockDim.x)
        Wt[k] = (k < K_) ? (_Float16)Ws[k] : (_Float16)0.0f;
}

// out[b,t] = relu(sum_k Ah[b,k]*Wh[t,k] + bias[t]).
// One wave computes one 16x16 tile of out, looping K in steps of 32 with
// v_wmma_f32_16x16x32_f16.  Fragment layouts per CDNA5 ISA 7.12.2:
//   A (16x32 f16): lane l -> row m=l&15, half=l>>4;
//     elems 0..7  = A[m, k0 + 8*half + 0..7]
//     elems 8..15 = A[m, k0 + 16 + 8*half + 0..7]
//   B (32x16 f16): lane l -> col n=l&15, khalf=l>>4;
//     elems 0..7  = B[k0 + 8*khalf + 0..7, n], elems 8..15 = B[k0 + 16 + 8*khalf + 0..7, n]  (== Wh[n, ...])
//   C/D (16x16 f32): lane l -> col n=l&15; vgpr r -> row r + 8*(l>>4)
__global__ __launch_bounds__(256)
void gemm_wmma_kernel(const _Float16* __restrict__ Ah, const _Float16* __restrict__ Wh,
                      const float* __restrict__ bias, float* __restrict__ out) {
    const int wid   = blockIdx.x * (blockDim.x >> 5) + (threadIdx.x >> 5);
    const int lane  = threadIdx.x & 31;
    const int mtile = wid >> 6;   // 0..63
    const int ntile = wid & 63;   // 0..63
    const int half  = lane >> 4;
    const int mn    = lane & 15;

    const _Float16* arow = Ah + (size_t)(mtile * 16 + mn) * KP_;
    const _Float16* brow = Wh + (size_t)(ntile * 16 + mn) * KP_;

    v8f acc = {};
    for (int k0 = 0; k0 < KP_; k0 += 32) {
        v8h a0 = *(const v8h*)(arow + k0 + half * 8);
        v8h a1 = *(const v8h*)(arow + k0 + 16 + half * 8);
        v16h afrag;
#pragma unroll
        for (int i = 0; i < 8; ++i) { afrag[i] = a0[i]; afrag[8 + i] = a1[i]; }

        v8h b0 = *(const v8h*)(brow + k0 + half * 8);
        v8h b1 = *(const v8h*)(brow + k0 + 16 + half * 8);
        v16h bfrag;
#pragma unroll
        for (int i = 0; i < 8; ++i) { bfrag[i] = b0[i]; bfrag[8 + i] = b1[i]; }

        acc = __builtin_amdgcn_wmma_f32_16x16x32_f16(
            /*neg_a=*/false, afrag, /*neg_b=*/false, bfrag,
            /*c_mod=*/(short)0, acc, /*reuse_a=*/false, /*reuse_b=*/false);
    }

    const int col     = ntile * 16 + mn;
    const int rowbase = mtile * 16 + half * 8;
    const float bv    = bias[col];
#pragma unroll
    for (int r = 0; r < 8; ++r) {
        float v = acc[r] + bv;
        out[(size_t)(rowbase + r) * DT_ + col] = fmaxf(v, 0.0f);
    }
}

extern "C" void kernel_launch(void* const* d_in, const int* in_sizes, int n_in,
                              void* d_out, int out_size, void* d_ws, size_t ws_size,
                              hipStream_t stream) {
    const float* jamo   = (const float*)d_in[0];
    const float* word   = (const float*)d_in[1];
    const float* entity = (const float*)d_in[2];
    const float* W      = (const float*)d_in[3];
    const float* bias   = (const float*)d_in[4];
    float* out          = (float*)d_out;

    char* ws = (char*)d_ws;
    int*      nz = (int*)(ws + OFF_NZ);
    float*    cj = (float*)(ws + OFF_CJ);
    float*    cw = (float*)(ws + OFF_CW);
    float*    ce = (float*)(ws + OFF_CE);
    _Float16* Ah = (_Float16*)(ws + OFF_AH);
    _Float16* Wh = (_Float16*)(ws + OFF_WH);

    init_nz_kernel<<<1, 64, 0, stream>>>(nz);

    colsum_kernel<<<B_, 256, 0, stream>>>(jamo,   cj, nz + 0, DJ_);
    colsum_kernel<<<B_, 256, 0, stream>>>(word,   cw, nz + 1, DW_);
    colsum_kernel<<<B_, 256, 0, stream>>>(entity, ce, nz + 2, DE_);

    pack_A_kernel<<<B_,  256, 0, stream>>>(cj, cw, ce, nz, Ah);
    pack_W_kernel<<<DT_, 256, 0, stream>>>(W, Wh);

    // 64x64 tiles of 16x16, 8 waves (256 threads) per block -> 512 blocks
    gemm_wmma_kernel<<<(64 * 64) / 8, 256, 0, stream>>>(Ah, Wh, bias, out);
}
